// MyMSA_57080115363900
// MI455X (gfx1250) — matrix operand order repaired and checked by reading, not compiled, not run on hardware
//
#include <hip/hip_runtime.h>
#include <hip/hip_bf16.h>

// ---------------------------------------------------------------------------
// MSA: per-head QKV projection + flash attention, WMMA f16 (f32 accumulate),
// async global->LDS staging of K/V chunks (double buffered).
// B=4, S=2048, D=1024, H=16, DH=64
// ---------------------------------------------------------------------------

typedef __attribute__((ext_vector_type(16))) _Float16 v16h;
typedef __attribute__((ext_vector_type(8)))  _Float16 h8;
typedef __attribute__((ext_vector_type(8)))  float    v8f;

static constexpr int Bsz = 4;
static constexpr int S   = 2048;
static constexpr int D   = 1024;
static constexpr int H   = 16;
static constexpr int DH  = 64;

// ---- fragment helpers -----------------------------------------------------
// A-matrix (16x32 f16): row M = lane%16.
//   lanes 0-15 : e=0..7 -> K=0..7,   e=8..15 -> K=16..23
//   lanes 16-31: e=0..7 -> K=8..15,  e=8..15 -> K=24..31   (hoff = 8*(lane/16))
__device__ inline v16h pack16(h8 lo, h8 hi) {
    v16h a;
#pragma unroll
    for (int i = 0; i < 8; ++i) { a[i] = lo[i]; a[i + 8] = hi[i]; }
    return a;
}
__device__ inline v16h load_a16(const _Float16* row, int kbase) {
    return pack16(*(const h8*)(row + kbase), *(const h8*)(row + kbase + 16));
}
// B-matrix (32x16 f16): col N = lane%16; same K order as A (p = base + 8*(lane/16)).
__device__ inline v16h load_b16(const _Float16* p) {
    return pack16(*(const h8*)p, *(const h8*)(p + 16));
}
__device__ inline v16h a_from_f32(const float* row, int kbase) {
    v16h a;
#pragma unroll
    for (int i = 0; i < 8; ++i) {
        a[i]     = (_Float16)row[kbase + i];
        a[i + 8] = (_Float16)row[kbase + 16 + i];
    }
    return a;
}
__device__ inline v16h b_from_f32(const float* p) {
    v16h b;
#pragma unroll
    for (int i = 0; i < 16; ++i) b[i] = (_Float16)p[i < 8 ? i : i + 8];
    return b;
}
__device__ inline v8f wmma_f16(v16h a, v16h b, v8f c) {
    return __builtin_amdgcn_wmma_f32_16x16x32_f16(false, a, false, b,
                                                  (short)0, c, false, false);
}
__device__ inline float rmax16(float v) {
#pragma unroll
    for (int m = 8; m > 0; m >>= 1) v = fmaxf(v, __shfl_xor(v, m, 16));
    return v;
}
__device__ inline float rsum16(float v) {
#pragma unroll
    for (int m = 8; m > 0; m >>= 1) v += __shfl_xor(v, m, 16);
    return v;
}

// ---- CDNA5 async global->LDS copy (ASYNCcnt-tracked, no VGPR round trip) ---
__device__ inline void async_ld128(unsigned lds_off, const void* gptr) {
    asm volatile("global_load_async_to_lds_b128 %0, %1, off"
                 :: "v"(lds_off), "v"(gptr) : "memory");
}
__device__ inline void wait_async0() {
    asm volatile("s_wait_asynccnt 0x0" ::: "memory");
}
// generic shared pointer -> wave-relative LDS byte offset (addr[31:0])
__device__ inline unsigned lds_off_of(const void* p) {
    return (unsigned)(uintptr_t)p;
}

// ---------------------------------------------------------------------------
// Kernel 1: per-head QKV projection. One wave = 16 rows x 64 out channels.
// Q,K stored row-major [bh][s][dh] f16 (Q pre-scaled by 1/sqrt(H)=0.25);
// V stored transposed [bh][dh][s] f16 so PV B-fragments are contiguous.
// ---------------------------------------------------------------------------
__device__ inline void proj_mat(v16h xa0, v16h xa1, const float* __restrict__ W,
                                const float* __restrict__ bias,
                                _Float16* __restrict__ dst, float scale,
                                bool transposed, int s0, int ln, int half) {
#pragma unroll
    for (int nt = 0; nt < 4; ++nt) {
        const int ecol = nt * 16 + ln;
        const float* wrow = W + (size_t)ecol * DH;     // W[e][*] == (W^T) col e
        v16h wb0 = b_from_f32(wrow + half * 8);       // d 0..31 slice
        v16h wb1 = b_from_f32(wrow + 32 + half * 8);  // d 32..63 slice
        const float bv = bias[ecol];
        v8f c;
#pragma unroll
        for (int r = 0; r < 8; ++r) c[r] = bv;
        c = wmma_f16(xa0, wb0, c);
        c = wmma_f16(xa1, wb1, c);
#pragma unroll
        for (int r = 0; r < 8; ++r) {
            const int srow = s0 + r + half * 8;        // C: M = r + 8*(lane/16)
            const _Float16 val = (_Float16)(c[r] * scale);
            if (transposed) dst[(size_t)ecol * S + srow] = val;
            else            dst[(size_t)srow * DH + ecol] = val;
        }
    }
}

__global__ __launch_bounds__(128) void msa_proj(
    const float* __restrict__ x,
    const float* __restrict__ Wq, const float* __restrict__ bq,
    const float* __restrict__ Wk, const float* __restrict__ bk,
    const float* __restrict__ Wv, const float* __restrict__ bv,
    _Float16* __restrict__ Qw, _Float16* __restrict__ Kw,
    _Float16* __restrict__ Vt) {
    const int bh   = blockIdx.x;
    const int b    = bh / H, h = bh % H;
    const int lane = threadIdx.x & 31;
    const int w    = threadIdx.x >> 5;
    const int ln   = lane & 15;
    const int half = lane >> 4;
    const int hoff = half * 8;
    const int s0   = blockIdx.y * 64 + w * 16;

    const float* xrow = x + ((size_t)b * S + (s0 + ln)) * D + (size_t)h * DH;
    v16h xa0 = a_from_f32(xrow, hoff);        // d = 0..31
    v16h xa1 = a_from_f32(xrow, 32 + hoff);   // d = 32..63

    const size_t hd2 = (size_t)h * DH * DH;
    proj_mat(xa0, xa1, Wq + hd2, bq + h * DH,
             Qw + (size_t)bh * S * DH, 0.25f, false, s0, ln, half);
    proj_mat(xa0, xa1, Wk + hd2, bk + h * DH,
             Kw + (size_t)bh * S * DH, 1.0f, false, s0, ln, half);
    proj_mat(xa0, xa1, Wv + hd2, bv + h * DH,
             Vt + (size_t)bh * DH * S, 1.0f, true, s0, ln, half);
}

// ---------------------------------------------------------------------------
// Kernel 2: flash attention. 4 waves/block share (b,h); each wave owns 16
// query rows. K/V chunks (32 keys) staged into LDS via async copies, double
// buffered; WMMAs consume LDS fragments while the next chunk streams in.
// ---------------------------------------------------------------------------
__global__ __launch_bounds__(128) void msa_attn(
    const _Float16* __restrict__ Qw, const _Float16* __restrict__ Kw,
    const _Float16* __restrict__ Vt, float* __restrict__ out) {
    __shared__ __align__(16) _Float16 Klds[2][32][DH];   // [buf][key][dh] 2x4KB
    __shared__ __align__(16) _Float16 Vlds[2][DH][32];   // [buf][dh][key] 2x4KB
    __shared__ __align__(16) _Float16 Plds[4][16][32];   // per-wave P tile 4KB

    const int tid  = threadIdx.x;
    const int bh   = blockIdx.x;
    const int lane = tid & 31;
    const int w    = tid >> 5;
    const int ln   = lane & 15;
    const int half = lane >> 4;
    const int hoff = half * 8;
    const int s0   = blockIdx.y * 64 + w * 16;

    const _Float16* qbase  = Qw + (size_t)bh * S * DH;
    const _Float16* kbase  = Kw + (size_t)bh * S * DH;
    const _Float16* vtbase = Vt + (size_t)bh * DH * S;

    const unsigned koff = lds_off_of(&Klds[0][0][0]);
    const unsigned voff = lds_off_of(&Vlds[0][0][0]);

    // Q fragments stay resident for the whole key loop
    const _Float16* qrow = qbase + (size_t)(s0 + ln) * DH;
    const v16h qa0 = load_a16(qrow, hoff);
    const v16h qa1 = load_a16(qrow, 32 + hoff);

    v8f o[4];
#pragma unroll
    for (int nt = 0; nt < 4; ++nt) o[nt] = (v8f){};
    float m[8], l[8];
#pragma unroll
    for (int r = 0; r < 8; ++r) { m[r] = -3.0e38f; l[r] = 0.0f; }

    // stage a 32-key chunk: K chunk is 4KB contiguous; Vt chunk is 64 rows
    // (dh) x 64B with stride S*2. 128 threads x 4 x 16B = 8KB per chunk.
    auto stage = [&](int c0, int buf) {
        const _Float16* kg = kbase + (size_t)c0 * DH + tid * 8;
        async_ld128(koff + buf * 4096 + tid * 16, kg);
        async_ld128(koff + buf * 4096 + 2048 + tid * 16, kg + 1024);
        const int vr = tid >> 2, vc = (tid & 3) * 8;
        async_ld128(voff + buf * 4096 + tid * 16,
                    vtbase + (size_t)vr * S + c0 + vc);
        async_ld128(voff + buf * 4096 + 2048 + tid * 16,
                    vtbase + (size_t)(vr + 32) * S + c0 + vc);
    };

    stage(0, 0);
    wait_async0();
    __syncthreads();

    constexpr int NCHUNK = S / 32;
    for (int ci = 0; ci < NCHUNK; ++ci) {
        const int cur = ci & 1;
        if (ci + 1 < NCHUNK) stage((ci + 1) * 32, cur ^ 1);

        // ---- scores: two 16x16 tiles over this 32-key chunk (Q pre-scaled)
        v8f sc[2];
#pragma unroll
        for (int t = 0; t < 2; ++t) {
            const _Float16* krow = &Klds[cur][t * 16 + ln][0];
            v16h kb0 = load_b16(krow + half * 8);        // dh 0..31
            v16h kb1 = load_b16(krow + 32 + half * 8);   // dh 32..63
            v8f c = (v8f){};
            c = wmma_f16(qa0, kb0, c);
            c = wmma_f16(qa1, kb1, c);
            sc[t] = c;
        }

        // ---- online softmax (rows r / 8+r live in this lane's half)
        float fac[8];
#pragma unroll
        for (int r = 0; r < 8; ++r) {
            float cmax = rmax16(fmaxf(sc[0][r], sc[1][r]));
            float mnew = fmaxf(m[r], cmax);
            float corr = __expf(m[r] - mnew);
            float p0 = __expf(sc[0][r] - mnew);
            float p1 = __expf(sc[1][r] - mnew);
            l[r] = l[r] * corr + rsum16(p0 + p1);
            m[r] = mnew;
            fac[r] = corr;
            sc[0][r] = p0;
            sc[1][r] = p1;
        }
#pragma unroll
        for (int nt = 0; nt < 4; ++nt)
#pragma unroll
            for (int r = 0; r < 8; ++r) o[nt][r] *= fac[r];

        // ---- relayout P (C layout -> A layout) through this wave's LDS tile
#pragma unroll
        for (int t = 0; t < 2; ++t)
#pragma unroll
            for (int r = 0; r < 8; ++r)
                Plds[w][r + half * 8][t * 16 + ln] = (_Float16)sc[t][r];
        const v16h pa = load_a16(&Plds[w][ln][0], hoff);   // K = key 0..31

        // ---- O += P @ V (4 N-tiles of dh); Vlds rows give contiguous frags
#pragma unroll
        for (int nt = 0; nt < 4; ++nt) {
            const _Float16* vp = &Vlds[cur][nt * 16 + ln][half * 8];
            o[nt] = wmma_f16(pa, load_b16(vp), o[nt]);
        }

        wait_async0();       // my share of next chunk landed in LDS
        __syncthreads();     // whole block done with cur + staging done
    }

    // ---- epilogue: normalize and scatter to [b][s][h*DH + dh] f32
    const int b = bh / H, h = bh % H;
    float inv[8];
#pragma unroll
    for (int r = 0; r < 8; ++r) inv[r] = 1.0f / l[r];
#pragma unroll
    for (int nt = 0; nt < 4; ++nt) {
#pragma unroll
        for (int r = 0; r < 8; ++r) {
            const int srow = s0 + r + half * 8;
            out[((size_t)b * S + srow) * D + (size_t)h * DH + nt * 16 + ln] =
                o[nt][r] * inv[r];
        }
    }
}

// ---------------------------------------------------------------------------
extern "C" void kernel_launch(void* const* d_in, const int* in_sizes, int n_in,
                              void* d_out, int out_size, void* d_ws,
                              size_t ws_size, hipStream_t stream) {
    const float* x  = (const float*)d_in[0];
    const float* Wq = (const float*)d_in[1];
    const float* bq = (const float*)d_in[2];
    const float* Wk = (const float*)d_in[3];
    const float* bk = (const float*)d_in[4];
    const float* Wv = (const float*)d_in[5];
    const float* bv = (const float*)d_in[6];
    float* out = (float*)d_out;

    const size_t per = (size_t)Bsz * H * S * DH;   // 8M f16 elements each
    _Float16* Qw = (_Float16*)d_ws;
    _Float16* Kw = Qw + per;
    _Float16* Vt = Kw + per;                        // total 48 MB of ws

    dim3 grid(Bsz * H, S / 64);                     // (64, 32)
    msa_proj<<<grid, 128, 0, stream>>>(x, Wq, bq, Wk, bk, Wv, bv, Qw, Kw, Vt);
    msa_attn<<<grid, 128, 0, stream>>>(Qw, Kw, Vt, out);
}
